// PrisonerDilemmaEngine_37752762532574
// MI455X (gfx1250) — matrix operand order repaired and checked by reading, not compiled, not run on hardware
//
#include <hip/hip_runtime.h>

// ---------------------------------------------------------------------------
// CDNA5 (gfx1250) implementation: bf16 WMMA for all GEMMs, f32 accumulate.
// ---------------------------------------------------------------------------

typedef __attribute__((ext_vector_type(16))) __bf16        v16bf;
typedef __attribute__((ext_vector_type(8)))  float         v8f;
typedef __attribute__((ext_vector_type(8)))  unsigned short u16x8;

#define HIDD   256
#define OUTD   128
#define MLPD   128
#define KCOMB  384
#define KMEM   160      // 129 zero-padded to 160 (5 WMMA K-steps)

__device__ __forceinline__ unsigned short f2bfu(float f) {
  unsigned u = __float_as_uint(f);
  u += 0x7FFFu + ((u >> 16) & 1u);              // round-to-nearest-even
  return (unsigned short)(u >> 16);
}
__device__ __forceinline__ float bfu2f(unsigned short s) {
  return __uint_as_float(((unsigned)s) << 16);
}

__device__ __forceinline__ v8f wmma_bf16(v16bf a, v16bf b, v8f c) {
  return __builtin_amdgcn_wmma_f32_16x16x32_bf16(false, a, false, b, (short)0, c,
                                                 false, false);
}

// A fragment (16x32 bf16): lane holds row (lane&15), 16 contiguous K values
// starting at k0 + (lane>>4)*16 (matches B fragment order).
__device__ __forceinline__ v16bf load_afrag(const unsigned short* base, int stride,
                                            int k0, int lane) {
  const int r = lane & 15, half = lane >> 4;
  const unsigned short* p = base + r * stride + k0 + half * 16;
  union { v16bf v; u16x8 h[2]; } t;
  t.h[0] = *(const u16x8*)(p);
  t.h[1] = *(const u16x8*)(p + 8);
  return t.v;
}
// B fragment (32x16 bf16): lane holds col (lane&15), 16 contiguous K values
// starting at kstep*32 + (lane>>4)*16. p is fully lane-resolved by caller.
__device__ __forceinline__ v16bf load_bfrag(const unsigned short* p) {
  union { v16bf v; u16x8 h[2]; } t;
  t.h[0] = *(const u16x8*)(p);
  t.h[1] = *(const u16x8*)(p + 8);
  return t.v;
}

// ---------------------------------------------------------------------------
// Prep: f32 -> bf16 weight conversion
// ---------------------------------------------------------------------------
__global__ void cvt_bf16(const float* __restrict__ src,
                         unsigned short* __restrict__ dst, int n) {
  int i = blockIdx.x * blockDim.x + threadIdx.x;
  if (i < n) dst[i] = f2bfu(src[i]);
}
__global__ void cvt_wih(const float* __restrict__ src,
                        unsigned short* __restrict__ dst) {
  int i = blockIdx.x * blockDim.x + threadIdx.x;
  if (i < 768 * KMEM) {
    int r = i / KMEM, c = i % KMEM;
    dst[i] = (c < 129) ? f2bfu(src[r * 129 + c]) : (unsigned short)0;
  }
}

// ---------------------------------------------------------------------------
// Pass 1: MLPs + GRU + payoff clamp.  8 waves/block, 16 cells/wave.
// ---------------------------------------------------------------------------
__global__ __launch_bounds__(256)
void pass1_kernel(const unsigned short* __restrict__ xb,
                  const float* __restrict__ hiddens,
                  const float* __restrict__ payoffs,
                  const unsigned short* __restrict__ a_w1b,
                  const unsigned short* __restrict__ g_w1b,
                  const unsigned short* __restrict__ a_w2b,
                  const unsigned short* __restrict__ g_w2b,
                  const unsigned short* __restrict__ wihb,
                  const unsigned short* __restrict__ whhb,
                  const float* __restrict__ a_b1, const float* __restrict__ a_b2,
                  const float* __restrict__ g_b1, const float* __restrict__ g_b2,
                  const float* __restrict__ bih, const float* __restrict__ bhh,
                  float* __restrict__ newh,
                  unsigned short* __restrict__ out_bf,
                  float* __restrict__ tension_g,
                  float* __restrict__ fsum, float* __restrict__ fsum_dc,
                  float* __restrict__ tsum_g, unsigned* __restrict__ tmax_g,
                  int fs, int dc) {
  extern __shared__ char smem[];
  const int tid  = threadIdx.x;
  const int wave = tid >> 5;
  const int lane = tid & 31;
  const int half = lane >> 4;
  const int n15  = lane & 15;
  const int cellbase = blockIdx.x * 128 + wave * 16;

  unsigned short* hidb  = (unsigned short*)(smem + wave * 24576);           // 16x256 bf16
  unsigned short* hid1a = (unsigned short*)(smem + wave * 24576 + 8192);    // 16x128 bf16
  unsigned short* hid1g = (unsigned short*)(smem + wave * 24576 + 12288);   // 16x128 bf16
  float*          outf  = (float*)        (smem + wave * 24576 + 16384);    // 16x128 f32
  unsigned short* memin = hid1a;                 // 16x160 bf16, reused after layer 2
  float* wgsum = (float*)(smem + 8 * 24576);     // 256 f32 (block h-sum per dim)
  float* tL    = (float*)(smem + 8 * 24576 + 1024);  // 128 f32 tensions

  // ---- stage hidden rows as bf16 ----
  wgsum[tid] = 0.f;
  for (int r = 0; r < 16; ++r)
    for (int c = lane; c < HIDD; c += 32)
      hidb[r * HIDD + c] = f2bfu(hiddens[(size_t)(cellbase + r) * HIDD + c]);
  __syncthreads();

  // ---- layer 1 (a & g share A fragments), K = 384 ----
  v8f accA[8], accG[8];
  v8f vz = {};
#pragma unroll
  for (int t = 0; t < 8; ++t) { accA[t] = vz; accG[t] = vz; }
  for (int ks = 0; ks < 12; ++ks) {
    v16bf A = (ks < 4) ? load_afrag(xb, 0, ks * 32, lane)          // x part: rows identical
                       : load_afrag(hidb, HIDD, ks * 32 - 128, lane);
    const int k0 = ks * 32 + half * 16;
#pragma unroll
    for (int nt = 0; nt < 8; ++nt) {
      const int nn = nt * 16 + n15;
      accA[nt] = wmma_bf16(A, load_bfrag(a_w1b + nn * KCOMB + k0), accA[nt]);
      accG[nt] = wmma_bf16(A, load_bfrag(g_w1b + nn * KCOMB + k0), accG[nt]);
    }
  }

  // ---- bias + relu -> bf16 LDS ----
#pragma unroll
  for (int nt = 0; nt < 8; ++nt) {
    const int nn = nt * 16 + n15;
    const float ba = a_b1[nn], bg = g_b1[nn];
#pragma unroll
    for (int v = 0; v < 8; ++v) {
      const int m = v + 8 * half;
      hid1a[m * MLPD + nn] = f2bfu(fmaxf(accA[nt][v] + ba, 0.f));
      hid1g[m * MLPD + nn] = f2bfu(fmaxf(accG[nt][v] + bg, 0.f));
    }
  }
  __syncthreads();

  // ---- layer 2, out = a2 - g2 + (a_b2 - g_b2), K = 128 ----
#pragma unroll
  for (int t = 0; t < 8; ++t) { accA[t] = vz; accG[t] = vz; }
  for (int ks = 0; ks < 4; ++ks) {
    v16bf Aa = load_afrag(hid1a, MLPD, ks * 32, lane);
    v16bf Ag = load_afrag(hid1g, MLPD, ks * 32, lane);
    const int k0 = ks * 32 + half * 16;
#pragma unroll
    for (int nt = 0; nt < 8; ++nt) {
      const int nn = nt * 16 + n15;
      accA[nt] = wmma_bf16(Aa, load_bfrag(a_w2b + nn * MLPD + k0), accA[nt]);
      accG[nt] = wmma_bf16(Ag, load_bfrag(g_w2b + nn * MLPD + k0), accG[nt]);
    }
  }
#pragma unroll
  for (int nt = 0; nt < 8; ++nt) {
    const int nn = nt * 16 + n15;
    const float bd = a_b2[nn] - g_b2[nn];
#pragma unroll
    for (int v = 0; v < 8; ++v)
      outf[(v + 8 * half) * OUTD + nn] = accA[nt][v] - accG[nt][v] + bd;
  }
  __syncthreads();

  // ---- tension + mem_in + out (bf16 to global) ----
  if (lane < 16) {
    const int r = lane;
    float s = 0.f;
    for (int c = 0; c < OUTD; ++c) { float o = outf[r * OUTD + c]; s += o * o; }
    const float t = s * (1.f / OUTD);
    tL[wave * 16 + r] = t;
    tension_g[cellbase + r] = t;
  }
  __syncthreads();
  if (tid == 0) {
    float mx = 0.f, sm = 0.f;
    for (int i = 0; i < 128; ++i) { float t = tL[i]; mx = fmaxf(mx, t); sm += t; }
    atomicMax(tmax_g, __float_as_uint(mx));   // tension >= 0: bit order == value order
    atomicAdd(tsum_g, sm);
  }
  for (int r = 0; r < 16; ++r) {
    const float tr = tL[wave * 16 + r];
    for (int c = lane; c < KMEM; c += 32) {
      float vv = (c < OUTD) ? outf[r * OUTD + c] : ((c == OUTD) ? tr : 0.f);
      unsigned short b = f2bfu(vv);
      memin[r * KMEM + c] = b;
      if (c < OUTD) out_bf[(size_t)(cellbase + r) * OUTD + c] = b;
    }
  }
  __syncthreads();

  // ---- GRU: 16 hidden-tiles, 6 accumulators each ----
  float pscale[8];
#pragma unroll
  for (int v = 0; v < 8; ++v)
    pscale[v] = 0.9f + 0.02f * payoffs[cellbase + v + 8 * half];

  for (int t = 0; t < 16; ++t) {
    const int d = t * 16 + n15;
    v8f ir = vz, iz = vz, in_ = vz, hr = vz, hz = vz, hn = vz;
    for (int ks = 0; ks < 5; ++ks) {               // gi: K = 160 (padded 129)
      v16bf A = load_afrag(memin, KMEM, ks * 32, lane);
      const int k0 = ks * 32 + half * 16;
      ir  = wmma_bf16(A, load_bfrag(wihb + (size_t)(      d) * KMEM + k0), ir);
      iz  = wmma_bf16(A, load_bfrag(wihb + (size_t)(256 + d) * KMEM + k0), iz);
      in_ = wmma_bf16(A, load_bfrag(wihb + (size_t)(512 + d) * KMEM + k0), in_);
    }
    for (int ks = 0; ks < 8; ++ks) {               // gh: K = 256
      v16bf A = load_afrag(hidb, HIDD, ks * 32, lane);
      const int k0 = ks * 32 + half * 16;
      hr = wmma_bf16(A, load_bfrag(whhb + (size_t)(      d) * HIDD + k0), hr);
      hz = wmma_bf16(A, load_bfrag(whhb + (size_t)(256 + d) * HIDD + k0), hz);
      hn = wmma_bf16(A, load_bfrag(whhb + (size_t)(512 + d) * HIDD + k0), hn);
    }
    const float bir = bih[d], biz = bih[256 + d], bin = bih[512 + d];
    const float bhr = bhh[d], bhz = bhh[256 + d], bhn = bhh[512 + d];
    float rs = 0.f;
#pragma unroll
    for (int v = 0; v < 8; ++v) {
      const int m = v + 8 * half;
      const int cell = cellbase + m;
      const float r_  = 1.f / (1.f + expf(-(ir[v] + bir + hr[v] + bhr)));
      const float z_  = 1.f / (1.f + expf(-(iz[v] + biz + hz[v] + bhz)));
      const float nn_ = tanhf(in_[v] + bin + r_ * (hn[v] + bhn));
      const float hold = hiddens[(size_t)cell * HIDD + d];
      float h = (1.f - z_) * nn_ + z_ * hold;
      h = fminf(10.f, fmaxf(-10.f, h * pscale[v]));
      newh[(size_t)cell * HIDD + d] = h;
      rs += h;
    }
    atomicAdd(&wgsum[d], rs);
  }
  __syncthreads();

  // ---- block partial sums -> faction accumulators ----
  {
    const int blk0 = blockIdx.x * 128;
    const int f = blk0 / fs;
    const bool indc = (blk0 % fs) < dc;
    const float v = wgsum[tid];
    atomicAdd(&fsum[f * HIDD + tid], v);
    if (indc) atomicAdd(&fsum_dc[f * HIDD + tid], v);
  }
}

// ---------------------------------------------------------------------------
// Stats: faction means, global mean M, exact post-debate mean gmean.
// ---------------------------------------------------------------------------
__global__ void stats_kernel(const float* __restrict__ fsum,
                             const float* __restrict__ fsum_dc,
                             const int* __restrict__ step_p,
                             float* __restrict__ fmean, float* __restrict__ Mv,
                             float* __restrict__ gmean, int n, int fs, int dc) {
  const int d = threadIdx.x;   // 256
  float tot = 0.f, sdc = 0.f;
  for (int f = 0; f < 8; ++f) {
    const float v = fsum[f * HIDD + d];
    fmean[f * HIDD + d] = v / (float)fs;
    tot += v;
    sdc += fsum_dc[f * HIDD + d];
  }
  const float M = tot / (float)n;
  Mv[d] = M;
  float g = M;
  if (step_p[0] > 5)   // sync preserves faction means -> closed form for post-debate mean
    g = M + 0.15f * 0.85f * (8.f * (float)dc * M - sdc) / (float)n;
  gmean[d] = g;
}

// ---------------------------------------------------------------------------
// Pass 2: sync + debate + coop pull + defector noise + clamp (in-place on newh)
// ---------------------------------------------------------------------------
__device__ __forceinline__ unsigned hashu(unsigned x) {
  x ^= x >> 16; x *= 0x7feb352dU; x ^= x >> 15; x *= 0x846ca68bU; x ^= x >> 16;
  return x;
}

__global__ void pass2_kernel(float* __restrict__ newh,
                             const int* __restrict__ last_action,
                             const int* __restrict__ step_p,
                             const float* __restrict__ fmean,
                             const float* __restrict__ Mv,
                             const float* __restrict__ gmean,
                             int n, int fs, int dc) {
  const size_t i = (size_t)blockIdx.x * blockDim.x + threadIdx.x;
  if (i >= (size_t)n * HIDD) return;
  const int cell  = (int)(i >> 8);
  const int d     = (int)(i & 255);
  const int f     = cell / fs;
  const int infac = cell % fs;
  float h = newh[i];
  h = 0.85f * h + 0.15f * fmean[f * HIDD + d];
  if (step_p[0] > 5 && infac < dc) h = 0.85f * h + 0.15f * Mv[d];
  const int la = last_action[cell];
  h += 0.05f * (float)la * (gmean[d] - h);
  if (la == 0) {
    const unsigned s  = (unsigned)i;
    const unsigned h1 = hashu(s * 2654435761u + 0x9e3779b9u);
    const unsigned h2 = hashu(h1 ^ 0x85ebca6bU);
    const float u1 = ((float)h1 + 0.5f) * (1.f / 4294967296.f);
    const float u2 = ((float)h2 + 0.5f) * (1.f / 4294967296.f);
    h += 0.02f * sqrtf(-2.f * logf(u1)) * cosf(6.28318530718f * u2);
  }
  newh[i] = fminf(10.f, fmaxf(-10.f, h));
}

// ---------------------------------------------------------------------------
// Softmax mixture: comb_num += sum_i exp(t_i - tmax) * out_i ; wsum += sum exp
// ---------------------------------------------------------------------------
__global__ void softmax_mix(const float* __restrict__ tension,
                            const unsigned short* __restrict__ out_bf,
                            const unsigned* __restrict__ tmax_g,
                            float* __restrict__ comb_num,
                            float* __restrict__ wsum_g, int n) {
  __shared__ float wsh[256];
  const int tid = threadIdx.x;
  const int c0  = blockIdx.x * 256;
  const float tmaxv = __uint_as_float(*tmax_g);
  wsh[tid] = (c0 + tid < n) ? expf(tension[c0 + tid] - tmaxv) : 0.f;
  __syncthreads();
  if (tid < 128) {
    float a = 0.f;
    for (int c = 0; c < 256; ++c)
      a += wsh[c] * bfu2f(out_bf[(size_t)(c0 + c) * OUTD + tid]);
    atomicAdd(&comb_num[tid], a);
  } else if (tid == 128) {
    float s = 0.f;
    for (int c = 0; c < 256; ++c) s += wsh[c];
    atomicAdd(wsum_g, s);
  }
}

// ---------------------------------------------------------------------------
// Finalize: pred = head_w @ (comb_num / wsum) + head_b ; avg_tension
// ---------------------------------------------------------------------------
__global__ void finalize_kernel(const float* __restrict__ comb_num,
                                const float* __restrict__ wsum_g,
                                const float* __restrict__ tsum_g,
                                const float* __restrict__ head_w,
                                const float* __restrict__ head_b,
                                float* __restrict__ pred,
                                float* __restrict__ avg_t, int n) {
  __shared__ float comb[128];
  const int tid = threadIdx.x;   // 128
  comb[tid] = comb_num[tid] / (*wsum_g);
  __syncthreads();
  float acc = head_b[tid];
  for (int o = 0; o < 128; ++o) acc += head_w[tid * 128 + o] * comb[o];
  pred[tid] = acc;
  if (tid == 0) *avg_t = (*tsum_g) / (float)n;
}

// ---------------------------------------------------------------------------
extern "C" void kernel_launch(void* const* d_in, const int* in_sizes, int n_in,
                              void* d_out, int out_size, void* d_ws, size_t ws_size,
                              hipStream_t stream) {
  (void)n_in; (void)out_size; (void)ws_size;
  const float* x        = (const float*)d_in[0];
  const float* payoffs  = (const float*)d_in[1];
  const int*   last_act = (const int*)  d_in[2];
  const int*   step_p   = (const int*)  d_in[3];
  const float* hiddens  = (const float*)d_in[4];
  const float* a_w1 = (const float*)d_in[5];
  const float* a_b1 = (const float*)d_in[6];
  const float* a_w2 = (const float*)d_in[7];
  const float* a_b2 = (const float*)d_in[8];
  const float* g_w1 = (const float*)d_in[9];
  const float* g_b1 = (const float*)d_in[10];
  const float* g_w2 = (const float*)d_in[11];
  const float* g_b2 = (const float*)d_in[12];
  const float* wih  = (const float*)d_in[13];
  const float* whh  = (const float*)d_in[14];
  const float* bih  = (const float*)d_in[15];
  const float* bhh  = (const float*)d_in[16];
  const float* head_w = (const float*)d_in[17];
  const float* head_b = (const float*)d_in[18];

  const int n  = in_sizes[1];           // N_CELLS
  const int fs = n / 8;                 // faction size
  const int dc = (fs / 4 > 1) ? fs / 4 : 1;

  float* pred  = (float*)d_out;         // [128]
  float* avg_t = pred + 128;            // [1]
  float* newh  = pred + 129;            // [n*256]

  // ---- workspace carve ----
  char* w = (char*)d_ws; size_t off = 0;
  auto carve = [&](size_t bytes) -> char* {
    char* p = w + off;
    off += (bytes + 255) & ~(size_t)255;
    return p;
  };
  unsigned short* xb     = (unsigned short*)carve(128 * 2);
  unsigned short* a_w1b  = (unsigned short*)carve(128 * 384 * 2);
  unsigned short* g_w1b  = (unsigned short*)carve(128 * 384 * 2);
  unsigned short* a_w2b  = (unsigned short*)carve(128 * 128 * 2);
  unsigned short* g_w2b  = (unsigned short*)carve(128 * 128 * 2);
  unsigned short* wihb   = (unsigned short*)carve(768 * KMEM * 2);
  unsigned short* whhb   = (unsigned short*)carve(768 * 256 * 2);
  unsigned short* out_bf = (unsigned short*)carve((size_t)n * 128 * 2);
  float* tension  = (float*)carve((size_t)n * 4);
  float* fsum     = (float*)carve(8 * 256 * 4);
  float* fsum_dc  = (float*)carve(8 * 256 * 4);
  float* fmean    = (float*)carve(8 * 256 * 4);
  float* Mv       = (float*)carve(256 * 4);
  float* gmean    = (float*)carve(256 * 4);
  float* comb_num = (float*)carve(128 * 4);
  float* scal     = (float*)carve(4 * 4);
  float*    tsum_g = scal;
  float*    wsum_g = scal + 1;
  unsigned* tmax_g = (unsigned*)(scal + 2);

  hipMemsetAsync(fsum,     0, 8 * 256 * 4, stream);
  hipMemsetAsync(fsum_dc,  0, 8 * 256 * 4, stream);
  hipMemsetAsync(comb_num, 0, 128 * 4,     stream);
  hipMemsetAsync(scal,     0, 4 * 4,       stream);

  // ---- weight conversion ----
  cvt_bf16<<<dim3(1),   dim3(128), 0, stream>>>(x,    xb,    128);
  cvt_bf16<<<dim3(192), dim3(256), 0, stream>>>(a_w1, a_w1b, 128 * 384);
  cvt_bf16<<<dim3(192), dim3(256), 0, stream>>>(g_w1, g_w1b, 128 * 384);
  cvt_bf16<<<dim3(64),  dim3(256), 0, stream>>>(a_w2, a_w2b, 128 * 128);
  cvt_bf16<<<dim3(64),  dim3(256), 0, stream>>>(g_w2, g_w2b, 128 * 128);
  cvt_bf16<<<dim3(768), dim3(256), 0, stream>>>(whh,  whhb,  768 * 256);
  cvt_wih<<<dim3((768 * KMEM + 255) / 256), dim3(256), 0, stream>>>(wih, wihb);

  // ---- pass 1 (heavy WMMA) ----
  const size_t smem = (size_t)8 * 24576 + 1024 + 512;   // ~194 KB (< 320 KB/WGP)
  hipFuncSetAttribute(reinterpret_cast<const void*>(pass1_kernel),
                      hipFuncAttributeMaxDynamicSharedMemorySize, (int)smem);
  pass1_kernel<<<dim3(n / 128), dim3(256), smem, stream>>>(
      xb, hiddens, payoffs, a_w1b, g_w1b, a_w2b, g_w2b, wihb, whhb,
      a_b1, a_b2, g_b1, g_b2, bih, bhh,
      newh, out_bf, tension, fsum, fsum_dc, tsum_g, tmax_g, fs, dc);

  // ---- cross-cell stats ----
  stats_kernel<<<dim3(1), dim3(256), 0, stream>>>(fsum, fsum_dc, step_p,
                                                  fmean, Mv, gmean, n, fs, dc);

  // ---- pass 2: sync/debate/coop/noise/clamp ----
  const size_t tot = (size_t)n * 256;
  pass2_kernel<<<dim3((unsigned)((tot + 255) / 256)), dim3(256), 0, stream>>>(
      newh, last_act, step_p, fmean, Mv, gmean, n, fs, dc);

  // ---- softmax mixture + head ----
  softmax_mix<<<dim3(n / 256), dim3(256), 0, stream>>>(tension, out_bf, tmax_g,
                                                       comb_num, wsum_g, n);
  finalize_kernel<<<dim3(1), dim3(128), 0, stream>>>(comb_num, wsum_g, tsum_g,
                                                     head_w, head_b, pred, avg_t, n);
}
